// ManifoldRNN_79628693668148
// MI455X (gfx1250) — compile-verified, matrix operand order checked by reading, not run on hardware
//
#include <hip/hip_runtime.h>
#include <hip/hip_bf16.h>

// ---------------------------------------------------------------------------
// 2-layer tanh RNN for MI455X (gfx1250, wave32, WMMA bf16 16x16x32).
//   Phase A (parallel): px[t] = x_t @ w_h0 + b_h0, stored as packed bf16
//                       C-fragments in d_ws.
//   Phase B (1 WGP, sequential): scan over t with weights resident in VGPRs
//                       (bf16 B-fragments) and h-state ping-ponged in LDS;
//                       exports h1n per step as A-fragments to d_ws.
//   Phase C (parallel): out[b,t,:] = h1n[t] @ w_q + b_q.
// ---------------------------------------------------------------------------

typedef __attribute__((ext_vector_type(16))) __bf16        bf16x16;
typedef __attribute__((ext_vector_type(2)))  __bf16        bf16x2;
typedef __attribute__((ext_vector_type(8)))  float         f32x8;
typedef __attribute__((ext_vector_type(8)))  unsigned int  u32x8;

#define S_LEN 2048
#define BATCH 64
#define HD    256
#define HTILE (BATCH * HD)    // one [64][256] bf16 state tile (shorts)

#if __has_builtin(__builtin_amdgcn_tanhf)
#define TANHF(x) __builtin_amdgcn_tanhf(x)
#else
#define TANHF(x) tanhf(x)
#endif

// Native f32 -> bf16 conversion (clang lowers to v_cvt_*bf16* on gfx1250).
static __device__ __forceinline__ unsigned short bf16c(float f) {
    __bf16 b = (__bf16)f;
    return __builtin_bit_cast(unsigned short, b);
}
static __device__ __forceinline__ unsigned pack2bf16(float lo, float hi) {
    bf16x2 v = { (__bf16)lo, (__bf16)hi };
    return __builtin_bit_cast(unsigned, v);
}
static __device__ __forceinline__ float bf16lo_f32(unsigned u) { return __uint_as_float(u << 16); }
static __device__ __forceinline__ float bf16hi_f32(unsigned u) { return __uint_as_float(u & 0xFFFF0000u); }

// A-fragment (16-bit A 16x32 layout, ISA 7.12.2) from a row-major bf16 [64][256]
// LDS tile: lanes 0-15 -> M=r,K=0..7/16..23 ; lanes 16-31 -> K+8. Two b128 reads.
static __device__ __forceinline__ bf16x16 lds_afrag(const unsigned short* h, int mt, int kf, int lane) {
    const int r = lane & 15, half = lane >> 4;
    const int row = mt * 16 + r;
    const int colbase = kf * 32 + half * 8;
    const uint4* p = (const uint4*)(h + row * HD + colbase);
    const uint4 lo = p[0];          // K = colbase .. colbase+7
    const uint4 hi = p[2];          // K = colbase+16 .. colbase+23
    u32x8 v = { lo.x, lo.y, lo.z, lo.w, hi.x, hi.y, hi.z, hi.w };
    return __builtin_bit_cast(bf16x16, v);
}

// B-fragment (mirror of A with N<->M roles) built from a row-major f32 weight
// W[K][256]: lane holds column n = nb*16 + (lane&15); dword d holds K pair.
static __device__ __forceinline__ u32x8 load_bfrag(const float* __restrict__ W, int nb, int kf, int lane) {
    const int r = lane & 15, half = lane >> 4;
    const int n  = nb * 16 + r;
    const int kb = kf * 32 + half * 8;
    u32x8 b;
#pragma unroll
    for (int d = 0; d < 8; ++d) {
        const int kk = (d < 4) ? (2 * d) : (16 + 2 * (d - 4));
        const int k  = kb + kk;
        b[d] = pack2bf16(W[(size_t)k * HD + n], W[(size_t)(k + 1) * HD + n]);
    }
    return b;
}

static __device__ __forceinline__ f32x8 wmma_bf16(bf16x16 a, u32x8 b, f32x8 c) {
    return __builtin_amdgcn_wmma_f32_16x16x32_bf16(
        /*neg_a=*/false, a, /*neg_b=*/false, __builtin_bit_cast(bf16x16, b),
        /*c_mod=*/(short)0, c, /*reuse_a=*/false, /*reuse_b=*/false);
}

// ---------------------------------------------------------------------------
// Phase A: px4[t][nb][mt][lane] = packed-bf16 C-fragment of (x_t @ w_h0 + b_h0)
// ---------------------------------------------------------------------------
__global__ __launch_bounds__(512, 1) void rnn_phaseA(
    const float* __restrict__ x, const float* __restrict__ w_h0,
    const float* __restrict__ b_h0, uint4* __restrict__ px4)
{
    __shared__ unsigned short ldsx[HTILE];            // 32 KB bf16 x_t tile
    const int t    = blockIdx.x;
    const int tid  = threadIdx.x;
    const int lane = tid & 31;
    const int nb   = tid >> 5;

    // stage x_t (fp32 global -> bf16 LDS), fully coalesced b128 reads
    {
        const int b  = tid >> 3;
        const int i0 = (tid & 7) * 32;
        const float4* src = (const float4*)(x + ((size_t)b * S_LEN + t) * HD);
#pragma unroll
        for (int j = 0; j < 8; ++j) {
            float4 v = src[(i0 >> 2) + j];
            uint2 pk = { pack2bf16(v.x, v.y), pack2bf16(v.z, v.w) };
            *(uint2*)(ldsx + b * HD + i0 + j * 4) = pk;
        }
    }

    u32x8 wf[8];
#pragma unroll
    for (int kf = 0; kf < 8; ++kf) wf[kf] = load_bfrag(w_h0, nb, kf, lane);
    const float bias = b_h0[nb * 16 + (lane & 15)];

    __syncthreads();

#pragma unroll
    for (int mt = 0; mt < 4; ++mt) {
        f32x8 acc;
#pragma unroll
        for (int r = 0; r < 8; ++r) acc[r] = bias;
#pragma unroll
        for (int kf = 0; kf < 8; ++kf)
            acc = wmma_bf16(lds_afrag(ldsx, mt, kf, lane), wf[kf], acc);
        uint4 q = { pack2bf16(acc[0], acc[1]), pack2bf16(acc[2], acc[3]),
                    pack2bf16(acc[4], acc[5]), pack2bf16(acc[6], acc[7]) };
        px4[(((size_t)t * 16 + nb) * 4 + mt) * 32 + lane] = q;
    }
}

// ---------------------------------------------------------------------------
// Phase B: sequential scan. 16 waves; wave nb owns output columns
// [nb*16, nb*16+16). All recurrent weights live in VGPRs as B-fragments.
// LDS layout (dynamic): [0]=h0 bufA, [1]=h0 bufB, [2]=h1 bufA, [3]=h1 bufB.
// ---------------------------------------------------------------------------
__global__ __launch_bounds__(512, 1) void rnn_phaseB(
    const uint4* __restrict__ px4,
    const float* __restrict__ u_h0, const float* __restrict__ w_h1,
    const float* __restrict__ u_h1, const float* __restrict__ b_h1,
    uint4* __restrict__ hfrag, float* __restrict__ out_tail)
{
    extern __shared__ unsigned short smem[];          // 4 x 32 KB ping-pong

    const int tid  = threadIdx.x;
    const int lane = tid & 31;
    const int nb   = tid >> 5;
    const int r16  = lane & 15, half = lane >> 4;
    const int col  = nb * 16 + r16;

    for (int i = tid; i < 4 * HTILE; i += 512) smem[i] = 0;   // h0 = h1 = 0

    u32x8 wu0[8], ww1[8], wu1[8];
#pragma unroll
    for (int kf = 0; kf < 8; ++kf) {
        wu0[kf] = load_bfrag(u_h0, nb, kf, lane);
        ww1[kf] = load_bfrag(w_h1, nb, kf, lane);
        wu1[kf] = load_bfrag(u_h1, nb, kf, lane);
    }
    const float bias1 = b_h1[col];

    __syncthreads();

    int cur = 0;
    for (int t = 0; t < S_LEN; ++t) {
        // ping-pong pointers via plain offset arithmetic (no pointer arrays!)
        const unsigned short* h0c = smem + (size_t)cur * HTILE;
        unsigned short*       h0n = smem + (size_t)(cur ^ 1) * HTILE;
        const unsigned short* h1c = smem + (size_t)(2 + cur) * HTILE;
        unsigned short*       h1n = smem + (size_t)(2 + (cur ^ 1)) * HTILE;

        // ---- stage 1: h0n = tanh(px[t] + h0c @ u_h0) ----
#pragma unroll
        for (int mt = 0; mt < 4; ++mt) {
            uint4 q = px4[(((size_t)t * 16 + nb) * 4 + mt) * 32 + lane];
            if (t + 1 < S_LEN)   // warm L2/WGP$ for the next step's fragment
                __builtin_prefetch(&px4[(((size_t)(t + 1) * 16 + nb) * 4 + mt) * 32 + lane], 0, 0);
            f32x8 acc;
            acc[0] = bf16lo_f32(q.x); acc[1] = bf16hi_f32(q.x);
            acc[2] = bf16lo_f32(q.y); acc[3] = bf16hi_f32(q.y);
            acc[4] = bf16lo_f32(q.z); acc[5] = bf16hi_f32(q.z);
            acc[6] = bf16lo_f32(q.w); acc[7] = bf16hi_f32(q.w);
#pragma unroll
            for (int kf = 0; kf < 8; ++kf)
                acc = wmma_bf16(lds_afrag(h0c, mt, kf, lane), wu0[kf], acc);
#pragma unroll
            for (int r = 0; r < 8; ++r) {
                float v = TANHF(acc[r]);
                h0n[(mt * 16 + r + 8 * half) * HD + col] = bf16c(v);
            }
        }
        __syncthreads();

        // ---- stage 2: h1n = tanh(b_h1 + h0n @ w_h1 + h1c @ u_h1) ----
#pragma unroll
        for (int mt = 0; mt < 4; ++mt) {
            f32x8 acc;
#pragma unroll
            for (int r = 0; r < 8; ++r) acc[r] = bias1;
#pragma unroll
            for (int kf = 0; kf < 8; ++kf)
                acc = wmma_bf16(lds_afrag(h0n, mt, kf, lane), ww1[kf], acc);
#pragma unroll
            for (int kf = 0; kf < 8; ++kf)
                acc = wmma_bf16(lds_afrag(h1c, mt, kf, lane), wu1[kf], acc);
#pragma unroll
            for (int r = 0; r < 8; ++r) {
                float v = TANHF(acc[r]);
                h1n[(mt * 16 + r + 8 * half) * HD + col] = bf16c(v);
            }
        }
        __syncthreads();

        // ---- export h1n as A-fragments (LDS does the transpose) ----
#pragma unroll
        for (int e = 0; e < 2; ++e) {
            const int f  = nb * 2 + e;          // 32 fragments, 2 per wave
            const int mt = f >> 3, kf = f & 7;
            const int row     = mt * 16 + r16;
            const int colbase = kf * 32 + half * 8;
            const uint4* p = (const uint4*)(h1n + row * HD + colbase);
            uint4* dst = hfrag + (((size_t)t * 32 + f) * 32 + lane) * 2;
            dst[0] = p[0];
            dst[1] = p[2];
        }
        cur ^= 1;
    }

    // final hidden states -> h_final [B, 2, H] fp32
    const unsigned short* h0f = smem + (size_t)cur * HTILE;
    const unsigned short* h1f = smem + (size_t)(2 + cur) * HTILE;
    for (int idx = tid; idx < BATCH * 2 * HD; idx += 512) {
        const int b  = idx >> 9;
        const int l  = (idx >> 8) & 1;
        const int hh = idx & 255;
        const unsigned short v = l ? h1f[b * HD + hh] : h0f[b * HD + hh];
        out_tail[idx] = __uint_as_float((unsigned)v << 16);
    }
}

// ---------------------------------------------------------------------------
// Phase C: out[b, t, :] = h1n[t] @ w_q + b_q  (A-fragments straight from ws)
// ---------------------------------------------------------------------------
__global__ __launch_bounds__(512, 1) void rnn_phaseC(
    const uint4* __restrict__ hfrag, const float* __restrict__ w_q,
    const float* __restrict__ b_q, float* __restrict__ out)
{
    const int t    = blockIdx.x;
    const int tid  = threadIdx.x;
    const int lane = tid & 31;
    const int nb   = tid >> 5;
    const int r16  = lane & 15, half = lane >> 4;

    u32x8 wf[8];
#pragma unroll
    for (int kf = 0; kf < 8; ++kf) wf[kf] = load_bfrag(w_q, nb, kf, lane);
    const int   col  = nb * 16 + r16;
    const float bias = b_q[col];

#pragma unroll
    for (int mt = 0; mt < 4; ++mt) {
        f32x8 acc;
#pragma unroll
        for (int rr = 0; rr < 8; ++rr) acc[rr] = bias;
#pragma unroll
        for (int kf = 0; kf < 8; ++kf) {
            const int f = mt * 8 + kf;
            const uint4* p = hfrag + (((size_t)t * 32 + f) * 32 + lane) * 2;
            const uint4 lo = p[0], hi = p[1];
            u32x8 av = { lo.x, lo.y, lo.z, lo.w, hi.x, hi.y, hi.z, hi.w };
            acc = wmma_bf16(__builtin_bit_cast(bf16x16, av), wf[kf], acc);
        }
#pragma unroll
        for (int rr = 0; rr < 8; ++rr) {
            const int row = mt * 16 + rr + 8 * half;           // batch index
            out[((size_t)row * S_LEN + t) * HD + col] = acc[rr];
        }
    }
}

// ---------------------------------------------------------------------------
extern "C" void kernel_launch(void* const* d_in, const int* in_sizes, int n_in,
                              void* d_out, int out_size, void* d_ws, size_t ws_size,
                              hipStream_t stream) {
    (void)in_sizes; (void)n_in; (void)out_size; (void)ws_size;

    const float* x    = (const float*)d_in[0];
    const float* w_h0 = (const float*)d_in[1];
    const float* u_h0 = (const float*)d_in[2];
    const float* b_h0 = (const float*)d_in[3];
    const float* w_h1 = (const float*)d_in[4];
    const float* u_h1 = (const float*)d_in[5];
    const float* b_h1 = (const float*)d_in[6];
    const float* w_q  = (const float*)d_in[7];
    const float* b_q  = (const float*)d_in[8];
    float* out = (float*)d_out;

    uint4* px4   = (uint4*)d_ws;                              // 64 MiB
    uint4* hfrag = px4 + (size_t)S_LEN * 16 * 4 * 32;         // +64 MiB
    float* out_tail = out + (size_t)BATCH * S_LEN * HD;       // h_final

    rnn_phaseA<<<dim3(S_LEN), dim3(512), 0, stream>>>(x, w_h0, b_h0, px4);
    rnn_phaseB<<<dim3(1), dim3(512), 4 * HTILE * sizeof(unsigned short), stream>>>(
        px4, u_h0, w_h1, u_h1, b_h1, hfrag, out_tail);
    rnn_phaseC<<<dim3(S_LEN), dim3(512), 0, stream>>>(hfrag, w_q, b_q, out);
}
